// StateGNNConv_83322365542771
// MI455X (gfx1250) — compile-verified, matrix operand order checked by reading, not run on hardware
//
#include <hip/hip_runtime.h>
#include <math.h>

// ---------------------------------------------------------------------------
// CDNA5 WMMA types (wave32): v_wmma_f32_16x16x32_bf16
// ---------------------------------------------------------------------------
typedef __attribute__((ext_vector_type(16))) __bf16 v16bf;
typedef __attribute__((ext_vector_type(8)))  __bf16 v8bf;   // 16 bytes
typedef __attribute__((ext_vector_type(2)))  __bf16 v2bf;   // 4 bytes
typedef __attribute__((ext_vector_type(8)))  float  v8f;
typedef __attribute__((ext_vector_type(4)))  int    vint4;  // native clang vector

#define D      128
#define ROWS   32        // node rows per GEMM block: 2 row-tiles x 8 col-tiles = 16 waves
#define NTHR   512
#define GMAX   8
#define SLOPE  0.01f
#define EPS    1e-6f

__device__ __forceinline__ float leakyf(float x) { return x >= 0.f ? x : SLOPE * x; }

// ---------------------------------------------------------------------------
// Swizzled fragment layout. For a [M x 128] bf16 tile (M multiple of 16):
// fragment f = (m>>4)*4 + (k>>5) holds a 16x32 WMMA operand. Within it,
// ISA layout (16-bit A/B): lane = (m&15) + ((k&8)?16:0),
// element j = (k&7) + ((k&16)?8:0). We store the two 8-element halves of each
// lane 256 elements apart so each half is a conflict-free ds_load_b128
// (lanes at 16B stride covering all 64 banks).
// ---------------------------------------------------------------------------
__device__ __host__ __forceinline__ int swz_off(int m, int k)
{
  return (((m >> 4) * 4 + ((k >> 5) & 3)) * 512)
       + ((k & 16) ? 256 : 0)
       + (((m & 15) + ((k & 8) ? 16 : 0)) * 8)
       + (k & 7);
}

__device__ __forceinline__ v16bf load_frag(const __bf16* s, int mblk, int kc, int lane)
{
  const __bf16* base = s + (mblk * 4 + kc) * 512 + lane * 8;
  union { v16bf v; v8bf h[2]; } u;
  u.h[0] = *(const v8bf*)(base);        // ds_load_b128
  u.h[1] = *(const v8bf*)(base + 256);  // ds_load_b128
  return u.v;
}

// One wave: C[mblk*16 : +16, nblk*16 : +16] += A(32x128) @ W(128x128), K=128
__device__ __forceinline__ v8f wave_gemm(const __bf16* sA, const __bf16* sW,
                                         int mblk, int nblk, int lane, v8f acc)
{
#pragma unroll
  for (int kc = 0; kc < 4; ++kc) {
    v16bf a = load_frag(sA, mblk, kc, lane);
    v16bf b = load_frag(sW, nblk, kc, lane);
    acc = __builtin_amdgcn_wmma_f32_16x16x32_bf16(false, a, false, b,
                                                  (short)0, acc, false, false);
  }
  return acc;
}

// ---------------------------------------------------------------------------
// Weight staging: global (pre-swizzled bf16) -> LDS, contiguous 16B chunks.
// Uses gfx1250 async global->LDS copy when available. Signature (from hipcc
// diagnostics): (native int4 AS1*, native int4 AS3*, imm offset, imm cpol).
// ---------------------------------------------------------------------------
#if __has_builtin(__builtin_amdgcn_global_load_async_to_lds_b128) && \
    __has_builtin(__builtin_amdgcn_s_wait_asynccnt)
#define HAVE_ASYNC_LDS 1
typedef __attribute__((address_space(1))) vint4 gvint4;
typedef __attribute__((address_space(3))) vint4 lvint4;
#else
#define HAVE_ASYNC_LDS 0
#endif

__device__ __forceinline__ void stage_weights(const __bf16* __restrict__ gW,
                                              __bf16* __restrict__ sW, int tid)
{
#if HAVE_ASYNC_LDS
  for (int i = tid; i < (D * D) / 8; i += NTHR) {
    __builtin_amdgcn_global_load_async_to_lds_b128(
        (gvint4*)(gW + i * 8), (lvint4*)(sW + i * 8), 0, 0);
  }
  __builtin_amdgcn_s_wait_asynccnt(0);
#else
  const vint4* src = (const vint4*)gW;
  vint4*       dst = (vint4*)sW;
  for (int i = tid; i < (D * D) / 8; i += NTHR)
    dst[i] = src[i];
#endif
}

// Activation staging: row-major f32 global -> swizzled bf16 LDS.
// k-pairs stay contiguous in the swizzle, so each thread stores packed 4B.
template <bool LEAKY>
__device__ __forceinline__ void stage_tile(const float* __restrict__ g, int row0, int N,
                                           __bf16* __restrict__ s, int tid)
{
  for (int i = tid; i < ROWS * D / 2; i += NTHR) {
    int e = i * 2;
    int r = e >> 7, k = e & 127;
    int gr = row0 + r;
    float v0 = 0.f, v1 = 0.f;
    if (gr < N) {
      float2 p = *(const float2*)&g[(long long)gr * D + k];
      v0 = p.x; v1 = p.y;
    }
    if (LEAKY) { v0 = leakyf(v0); v1 = leakyf(v1); }
    v2bf pk; pk[0] = (__bf16)v0; pk[1] = (__bf16)v1;
    *(v2bf*)&s[swz_off(r, k)] = pk;   // (k even) -> 4B aligned
  }
}

// ---------------------------------------------------------------------------
// Kernel 0: convert the four 128x128 fp32 weights to bf16, PRE-SWIZZLED into
// fragment layout in ws: [W1 | W2 | Wo | Wn], each D*D elements.
// (n plays the "m" role of the B operand; k is the contraction dim.)
// ---------------------------------------------------------------------------
__global__ void convert_weights_kernel(const float* __restrict__ W1,
                                       const float* __restrict__ W2,
                                       const float* __restrict__ Wo,
                                       const float* __restrict__ Wn,
                                       __bf16* __restrict__ Wb)
{
  int i = blockIdx.x * blockDim.x + threadIdx.x;   // 0 .. 4*D*D-1
  if (i >= 4 * D * D) return;
  int which = i >> 14;
  int idx   = i & (D * D - 1);
  int k = idx >> 7;           // row of W = contraction index
  int n = idx & 127;          // col of W = output channel
  const float* src = (which == 0) ? W1 : (which == 1) ? W2 : (which == 2) ? Wo : Wn;
  Wb[which * D * D + swz_off(n, k)] = (__bf16)src[idx];
}

// ---------------------------------------------------------------------------
// Zero a float range (grid-stride)
// ---------------------------------------------------------------------------
__global__ void zero_f32_kernel(float* __restrict__ p, long long n)
{
  long long stride = (long long)gridDim.x * blockDim.x;
  for (long long i = blockIdx.x * (long long)blockDim.x + threadIdx.x; i < n; i += stride)
    p[i] = 0.f;
}

// ---------------------------------------------------------------------------
// Kernel 1: h = leaky(x @ W1) @ W2
// ---------------------------------------------------------------------------
__global__ __launch_bounds__(NTHR) void mlp_kernel(const float* __restrict__ x,
                                                   const __bf16* __restrict__ Wb,
                                                   float* __restrict__ h, int N)
{
  __shared__ __attribute__((aligned(16))) __bf16 sW[D * D];      // 32 KB
  __shared__ __attribute__((aligned(16))) __bf16 sA[ROWS * D];   // 8 KB
  __shared__ __attribute__((aligned(16))) __bf16 sT[ROWS * D];   // 8 KB

  const int tid  = threadIdx.x;
  const int row0 = blockIdx.x * ROWS;

  stage_weights(Wb, sW, tid);                       // W1
  stage_tile<false>(x, row0, N, sA, tid);
  __syncthreads();

  const int lane = tid & 31;
  const int wave = tid >> 5;            // 0..15
  const int mblk = wave >> 3;           // 0 or 1
  const int nblk = wave & 7;            // 0..7
  const int nr   = nblk * 16 + (lane & 15);
  const int mb   = mblk * 16 + ((lane & 16) ? 8 : 0);

  v8f acc = {};
  acc = wave_gemm(sA, sW, mblk, nblk, lane, acc);

  // fused leaky -> swizzled intermediate tile
#pragma unroll
  for (int v = 0; v < 8; ++v)
    sT[swz_off(mb + v, nr)] = (__bf16)leakyf(acc[v]);
  __syncthreads();

  stage_weights(Wb + D * D, sW, tid);               // W2
  __syncthreads();

  v8f acc2 = {};
  acc2 = wave_gemm(sT, sW, mblk, nblk, lane, acc2);

#pragma unroll
  for (int v = 0; v < 8; ++v) {
    int gr = row0 + mb + v;
    if (gr < N) h[(long long)gr * D + nr] = acc2[v];
  }
}

// ---------------------------------------------------------------------------
// Kernel 2: edge gather/scale/scatter-sum: xn[dst,:] += w[e] * h[src,:]
// ---------------------------------------------------------------------------
__global__ __launch_bounds__(256) void edge_scatter_kernel(const int* __restrict__ ei,
                                                           const float* __restrict__ w,
                                                           const float* __restrict__ h,
                                                           float* __restrict__ xn, int E)
{
  int e = blockIdx.x * 2 + (threadIdx.x >> 7);
  int d = threadIdx.x & 127;
  if (e >= E) return;
  int dst = ei[e];        // edge_index[0][e]
  int src = ei[E + e];    // edge_index[1][e]
  float val = w[e] * h[(long long)src * D + d];
  atomicAdd(&xn[(long long)dst * D + d], val);
}

// ---------------------------------------------------------------------------
// Kernel 3: out = h @ Wo + leaky(xn) @ Wn   (accumulator chained across GEMMs)
// ---------------------------------------------------------------------------
__global__ __launch_bounds__(NTHR) void out_kernel(const float* __restrict__ h,
                                                   const float* __restrict__ xn,
                                                   const __bf16* __restrict__ Wb,
                                                   float* __restrict__ out, int N)
{
  __shared__ __attribute__((aligned(16))) __bf16 sW[D * D];
  __shared__ __attribute__((aligned(16))) __bf16 sH[ROWS * D];
  __shared__ __attribute__((aligned(16))) __bf16 sX[ROWS * D];

  const int tid  = threadIdx.x;
  const int row0 = blockIdx.x * ROWS;

  stage_weights(Wb + 2 * D * D, sW, tid);           // Wo
  stage_tile<false>(h,  row0, N, sH, tid);
  stage_tile<true >(xn, row0, N, sX, tid);          // fused leaky on aggregate
  __syncthreads();

  const int lane = tid & 31;
  const int wave = tid >> 5;
  const int mblk = wave >> 3;
  const int nblk = wave & 7;
  const int nr   = nblk * 16 + (lane & 15);
  const int mb   = mblk * 16 + ((lane & 16) ? 8 : 0);

  v8f acc = {};
  acc = wave_gemm(sH, sW, mblk, nblk, lane, acc);
  __syncthreads();

  stage_weights(Wb + 3 * D * D, sW, tid);           // Wn
  __syncthreads();

  acc = wave_gemm(sX, sW, mblk, nblk, lane, acc);   // chained C accumulate

#pragma unroll
  for (int v = 0; v < 8; ++v) {
    int gr = row0 + mb + v;
    if (gr < N) out[(long long)gr * D + nr] = acc[v];
  }
}

// ---------------------------------------------------------------------------
// GraphNorm stats pass 1: per-(graph,channel) sums.
// ---------------------------------------------------------------------------
__global__ __launch_bounds__(128) void stat_sum_kernel(const float* __restrict__ out,
                                                       const int* __restrict__ batch,
                                                       float* __restrict__ gsum, int N)
{
  int col  = threadIdx.x;
  int row0 = blockIdx.x * 64;
  if (row0 >= N) return;
  int end  = row0 + 64; if (end > N) end = N;
  int gcur = batch[row0];
  float acc = 0.f;
  for (int r = row0; r < end; ++r) {
    int g = batch[r];
    if (g != gcur) { atomicAdd(&gsum[gcur * D + col], acc); acc = 0.f; gcur = g; }
    acc += out[(long long)r * D + col];
  }
  atomicAdd(&gsum[gcur * D + col], acc);
}

// ---------------------------------------------------------------------------
// GraphNorm stats pass 2: per-(graph,channel) centered sum of squares.
// ---------------------------------------------------------------------------
__global__ __launch_bounds__(128) void stat_var_kernel(const float* __restrict__ out,
                                                       const int* __restrict__ batch,
                                                       const int* __restrict__ batch_num,
                                                       const float* __restrict__ gsum,
                                                       float* __restrict__ gss, int N)
{
  int col  = threadIdx.x;
  int row0 = blockIdx.x * 64;
  if (row0 >= N) return;
  int end  = row0 + 64; if (end > N) end = N;
  int gcur = batch[row0];
  float mu = gsum[gcur * D + col] / (float)batch_num[gcur];
  float acc = 0.f;
  for (int r = row0; r < end; ++r) {
    int g = batch[r];
    if (g != gcur) {
      atomicAdd(&gss[gcur * D + col], acc);
      acc = 0.f; gcur = g;
      mu = gsum[g * D + col] / (float)batch_num[g];
    }
    float dv = out[(long long)r * D + col] - mu;
    acc += dv * dv;
  }
  atomicAdd(&gss[gcur * D + col], acc);
}

// ---------------------------------------------------------------------------
// Finalize: y = (out - mu) / (sigma + eps) * gamma + beta
// ---------------------------------------------------------------------------
__global__ __launch_bounds__(256) void finalize_kernel(const float* __restrict__ out,
                                                       const int* __restrict__ batch,
                                                       const int* __restrict__ batch_num,
                                                       const float* __restrict__ gsum,
                                                       const float* __restrict__ gss,
                                                       const float* __restrict__ gamma,
                                                       const float* __restrict__ beta,
                                                       float* __restrict__ y, int N)
{
  long long total  = (long long)N * D;
  long long stride = (long long)gridDim.x * blockDim.x;
  for (long long idx = blockIdx.x * (long long)blockDim.x + threadIdx.x;
       idx < total; idx += stride) {
    int n = (int)(idx >> 7);
    int d = (int)(idx & 127);
    int g = batch[n];
    float cnt = (float)batch_num[g];
    float mu  = gsum[g * D + d] / cnt;
    float var = gss[g * D + d] / (cnt - 1.f);
    float sig = sqrtf(var);
    y[idx] = (out[idx] - mu) / (sig + EPS) * gamma[d] + beta[d];
  }
}

// ---------------------------------------------------------------------------
// Launcher
// Inputs: 0:x 1:edge_index 2:w 3:batch 4:batch_num 5:W1 6:W2 7:Wo 8:Wn
//         9:gamma 10:beta
// ---------------------------------------------------------------------------
extern "C" void kernel_launch(void* const* d_in, const int* in_sizes, int n_in,
                              void* d_out, int out_size, void* d_ws, size_t ws_size,
                              hipStream_t stream)
{
  const float* x      = (const float*)d_in[0];
  const int*   ei     = (const int*)  d_in[1];
  const float* w      = (const float*)d_in[2];
  const int*   batch  = (const int*)  d_in[3];
  const int*   bnum   = (const int*)  d_in[4];
  const float* W1     = (const float*)d_in[5];
  const float* W2     = (const float*)d_in[6];
  const float* Wo     = (const float*)d_in[7];
  const float* Wn     = (const float*)d_in[8];
  const float* gamma  = (const float*)d_in[9];
  const float* beta   = (const float*)d_in[10];

  const int N = in_sizes[0] / D;      // 40000
  const int E = in_sizes[2];          // 640000

  // Workspace layout (floats): [h | out | xn | gsum | gss | Wb(bf16, swizzled)]
  const long long ND = (long long)N * D;
  float*  ws   = (float*)d_ws;
  float*  h    = ws;
  float*  outb = ws + ND;
  float*  xn   = ws + 2 * ND;
  float*  gsum = ws + 3 * ND;                 // GMAX*D
  float*  gss  = gsum + GMAX * D;
  __bf16* Wb   = (__bf16*)(gss + GMAX * D);   // 4*D*D bf16, 16B-aligned

  convert_weights_kernel<<<(4 * D * D + 255) / 256, 256, 0, stream>>>(W1, W2, Wo, Wn, Wb);

  zero_f32_kernel<<<2048, 256, 0, stream>>>(xn, ND + 2LL * GMAX * D);

  mlp_kernel<<<(N + ROWS - 1) / ROWS, NTHR, 0, stream>>>(x, Wb, h, N);

  edge_scatter_kernel<<<(E + 1) / 2, 256, 0, stream>>>(ei, w, h, xn, E);

  out_kernel<<<(N + ROWS - 1) / ROWS, NTHR, 0, stream>>>(h, xn, Wb, outb, N);

  stat_sum_kernel<<<(N + 63) / 64, 128, 0, stream>>>(outb, batch, gsum, N);
  stat_var_kernel<<<(N + 63) / 64, 128, 0, stream>>>(outb, batch, bnum, gsum, gss, N);

  finalize_kernel<<<2048, 256, 0, stream>>>(outb, batch, bnum, gsum, gss,
                                            gamma, beta, (float*)d_out, N);
}
